// SpatialTransformer_4234837753923
// MI455X (gfx1250) — matrix operand order repaired and checked by reading, not compiled, not run on hardware
//
#include <hip/hip_runtime.h>

// ---------------------------------------------------------------------------
// SpatialTransformer for MI455X (gfx1250): KNN -> edge features -> 3-layer MLP
// (two big GEMMs via v_wmma_f32_16x16x32_f16, 32x64 register tile per wave)
// -> mean pool -> 3x3 transform.
// ---------------------------------------------------------------------------

typedef __attribute__((ext_vector_type(16))) _Float16 v16h;
typedef __attribute__((ext_vector_type(8)))  _Float16 v8h;
typedef __attribute__((ext_vector_type(8)))  float    v8f;

static constexpr int NPTS    = 65536;   // 32 * 2048
static constexpr int BGRAPH  = 32;
static constexpr int MPTS    = 2048;
static constexpr int KNN     = 16;
static constexpr int F1      = 96;      // K*6 input features
static constexpr int H1      = 1024;
static constexpr int H2      = 256;
static constexpr int CHUNK   = 8192;    // rows processed per pipeline pass
static constexpr int NCHUNK  = NPTS / CHUNK;

// ------------------------------- KNN ---------------------------------------
__global__ __launch_bounds__(256) void knn_kernel(const float* __restrict__ pos,
                                                  int* __restrict__ nbr) {
    __shared__ float sx[256], sy[256], sz[256];
    const int tid  = threadIdx.x;
    const int n    = blockIdx.x * 256 + tid;         // 2048 % 256 == 0 -> one graph per block
    const int base = (n >> 11) << 11;                // start of this graph

    const float px = pos[n * 3 + 0];
    const float py = pos[n * 3 + 1];
    const float pz = pos[n * 3 + 2];

    float bd[KNN];
    int   bi[KNN];
#pragma unroll
    for (int k = 0; k < KNN; ++k) { bd[k] = 3.0e38f; bi[k] = base; }

    for (int t = 0; t < MPTS; t += 256) {
        const int j = base + t + tid;
        sx[tid] = pos[j * 3 + 0];
        sy[tid] = pos[j * 3 + 1];
        sz[tid] = pos[j * 3 + 2];
        __syncthreads();
#pragma unroll 4
        for (int jj = 0; jj < 256; ++jj) {
            const float dx = px - sx[jj];
            const float dy = py - sy[jj];
            const float dz = pz - sz[jj];
            const float d2 = dx * dx + dy * dy + dz * dz;
            if (d2 < bd[KNN - 1]) {
                const int  jg    = base + t + jj;
                const bool front = d2 < bd[0];
#pragma unroll
                for (int k = KNN - 1; k >= 1; --k) {
                    const bool  shift = bd[k - 1] > d2;
                    const float cur   = bd[k];
                    const int   curi  = bi[k];
                    bd[k] = shift ? bd[k - 1] : ((cur > d2) ? d2 : cur);
                    bi[k] = shift ? bi[k - 1] : ((cur > d2) ? jg : curi);
                }
                if (front) { bd[0] = d2; bi[0] = jg; }
            }
        }
        __syncthreads();
    }
#pragma unroll
    for (int k = 0; k < KNN; ++k) nbr[(size_t)n * KNN + k] = bi[k];
}

// --------------------------- edge features ---------------------------------
// feat[r, k*6 + {0..5}] = [cx, cy, cz, cx-nx, cy-ny, cz-nz]  (f16, row-major)
__global__ __launch_bounds__(256) void feat_kernel(const float* __restrict__ pos,
                                                   const int* __restrict__ nbr,
                                                   _Float16* __restrict__ feat,
                                                   int rowStart) {
    const int r = blockIdx.x * 256 + threadIdx.x;    // 0..CHUNK-1
    const int n = rowStart + r;
    const float cx = pos[n * 3 + 0];
    const float cy = pos[n * 3 + 1];
    const float cz = pos[n * 3 + 2];
    _Float16* f = feat + (size_t)r * F1;
#pragma unroll
    for (int k = 0; k < KNN; ++k) {
        const int j = nbr[(size_t)n * KNN + k];
        const float nx = pos[j * 3 + 0];
        const float ny = pos[j * 3 + 1];
        const float nz = pos[j * 3 + 2];
        union { _Float16 h[6]; unsigned int u[3]; } t;
        t.h[0] = (_Float16)cx;
        t.h[1] = (_Float16)cy;
        t.h[2] = (_Float16)cz;
        t.h[3] = (_Float16)(cx - nx);
        t.h[4] = (_Float16)(cy - ny);
        t.h[5] = (_Float16)(cz - nz);
        unsigned int* dst = (unsigned int*)(f + k * 6);   // 12B per k, 4B aligned
        dst[0] = t.u[0]; dst[1] = t.u[1]; dst[2] = t.u[2];
    }
}

// --------------------- weight packing into B fragments ----------------------
// B-matrix (K x 16) fragments in this file's contiguous k order (same as A below):
//   lane L holds column N = L%16, K = 16*(L/16) + h for half h = 0..15.
// Wp[colTile][kf][lane][h]; each lane's fragment is 32 contiguous bytes.
__global__ __launch_bounds__(32) void pack_b_kernel(const float* __restrict__ W,
                                                    _Float16* __restrict__ Wp,
                                                    int Kdim, int Ncols) {
    const int colTile = blockIdx.x;
    const int kf      = blockIdx.y;
    const int lane    = threadIdx.x;
    const int kfN     = Kdim >> 5;
    _Float16* dst = Wp + ((size_t)colTile * kfN + kf) * 512 + lane * 16;
    const int col   = colTile * 16 + (lane & 15);
    const int krow0 = kf * 32 + ((lane >> 4) << 4);
#pragma unroll
    for (int h = 0; h < 16; ++h)
        dst[h] = (_Float16)W[(size_t)(krow0 + h) * Ncols + col];
}

// ------------------------------ WMMA GEMM ----------------------------------
// 32x64 output tile per wave: 2 row-tiles x 4 col-tiles = 8 independent v8f
// accumulators. Per K-step: 4 A-fragment loads + 8 B-fragment loads feed
// 8 v_wmma_f32_16x16x32_f16 (1.5 b128 loads per WMMA).
// A-fragment (16x32 f16): lane L -> row M = L%16,
//   halves 0..7  : K = kBase + 16*(L/16) + {0..7}      (one 16B load)
//   halves 8..15 : K = kBase + 16*(L/16) + 8 + {0..7}  (one 16B load)
// C/D: VGPR r, lane L -> M = r + 8*(L/16), N = L%16.
__global__ __launch_bounds__(128) void wmma_gemm_bias_relu(
        const _Float16* __restrict__ A, const _Float16* __restrict__ Wp,
        const float* __restrict__ bias, _Float16* __restrict__ Out,
        int Kdim, int Ncols, int doRelu) {
    const int lane     = threadIdx.x & 31;
    const int wave     = threadIdx.x >> 5;
    const int rowBase  = blockIdx.x * 32;             // 2 row tiles per wave
    const int colGroup = blockIdx.y * 4 + wave;       // 4 col tiles per wave
    const int colBase  = colGroup * 64;
    const int kfN      = Kdim >> 5;
    const int g        = lane >> 4;
    const int nloc     = lane & 15;

    const _Float16* ap0 = A + (size_t)(rowBase + nloc) * Kdim + (g << 4);
    const _Float16* ap1 = ap0 + (size_t)16 * Kdim;
    const _Float16* bp  = Wp + (size_t)(colGroup * 4) * kfN * 512 + lane * 16;

    v8f acc[8];                                       // acc[ct*2 + rt]
#pragma unroll
    for (int i = 0; i < 8; ++i) acc[i] = (v8f){};

    for (int kf = 0; kf < kfN; ++kf) {
        // Two A fragments (rows rowBase..+15 and rowBase+16..+31)
        const v8h a0lo = *(const v8h*)(ap0 + kf * 32);
        const v8h a0hi = *(const v8h*)(ap0 + kf * 32 + 8);
        const v8h a1lo = *(const v8h*)(ap1 + kf * 32);
        const v8h a1hi = *(const v8h*)(ap1 + kf * 32 + 8);
        v16h a[2];
#pragma unroll
        for (int i = 0; i < 8; ++i) {
            a[0][i] = a0lo[i]; a[0][i + 8] = a0hi[i];
            a[1][i] = a1lo[i]; a[1][i + 8] = a1hi[i];
        }
        // Four B fragments (col tiles colGroup*4 .. +3)
        v16h b[4];
#pragma unroll
        for (int ct = 0; ct < 4; ++ct) {
            const _Float16* bt = bp + ((size_t)ct * kfN + kf) * 512;
            const v8h blo = *(const v8h*)(bt);
            const v8h bhi = *(const v8h*)(bt + 8);
#pragma unroll
            for (int i = 0; i < 8; ++i) { b[ct][i] = blo[i]; b[ct][i + 8] = bhi[i]; }
        }
        // 8 independent WMMAs (no D->A/B hazards, full matrix-pipe ILP)
#pragma unroll
        for (int ct = 0; ct < 4; ++ct)
#pragma unroll
            for (int rt = 0; rt < 2; ++rt)
                acc[ct * 2 + rt] = __builtin_amdgcn_wmma_f32_16x16x32_f16(
                    false, a[rt], false, b[ct], (short)0, acc[ct * 2 + rt],
                    false, false);
    }

#pragma unroll
    for (int ct = 0; ct < 4; ++ct) {
        const int   col = colBase + ct * 16 + nloc;
        const float bv  = bias[col];
#pragma unroll
        for (int rt = 0; rt < 2; ++rt) {
#pragma unroll
            for (int r = 0; r < 8; ++r) {
                float v = acc[ct * 2 + rt][r] + bv;
                if (doRelu) v = v > 0.f ? v : 0.f;
                Out[(size_t)(rowBase + rt * 16 + r + 8 * g) * Ncols + col] =
                    (_Float16)v;
            }
        }
    }
}

// --------------------------- layer 3 (256 -> 9) ----------------------------
__global__ __launch_bounds__(256) void layer3_kernel(const _Float16* __restrict__ h2,
                                                     const float* __restrict__ W3,
                                                     const float* __restrict__ b3,
                                                     float* __restrict__ x9,
                                                     int rowStart) {
    const int r = blockIdx.x * 256 + threadIdx.x;
    const int n = rowStart + r;
    const v8h* hp = (const v8h*)(h2 + (size_t)r * H2);   // rows are 512B apart
    float acc[9];
#pragma unroll
    for (int c = 0; c < 9; ++c) acc[c] = b3[c];
    for (int i8 = 0; i8 < H2 / 8; ++i8) {
        const v8h hv = hp[i8];
#pragma unroll
        for (int j = 0; j < 8; ++j) {
            const float h = (float)hv[j];
            const float* w = W3 + (size_t)(i8 * 8 + j) * 9;
#pragma unroll
            for (int c = 0; c < 9; ++c) acc[c] += h * w[c];
        }
    }
#pragma unroll
    for (int c = 0; c < 9; ++c) x9[(size_t)n * 9 + c] = acc[c];
}

// ------------------------- per-graph mean pooling --------------------------
__global__ __launch_bounds__(256) void pool_kernel(const float* __restrict__ x9,
                                                   float* __restrict__ pooled) {
    __shared__ float s[9 * 256];
    const int b = blockIdx.x, tid = threadIdx.x;
    float acc[9];
#pragma unroll
    for (int c = 0; c < 9; ++c) acc[c] = 0.f;
    for (int r = tid; r < MPTS; r += 256) {
        const float* xp = x9 + ((size_t)b * MPTS + r) * 9;
#pragma unroll
        for (int c = 0; c < 9; ++c) acc[c] += xp[c];
    }
#pragma unroll
    for (int c = 0; c < 9; ++c) s[c * 256 + tid] = acc[c];
    __syncthreads();
    for (int off = 128; off > 0; off >>= 1) {
        if (tid < off)
#pragma unroll
            for (int c = 0; c < 9; ++c) s[c * 256 + tid] += s[c * 256 + tid + off];
        __syncthreads();
    }
    if (tid < 9) pooled[b * 9 + tid] = s[tid * 256] * (1.0f / (float)MPTS);
}

// -------------------- final per-point 3x3 transform ------------------------
__global__ __launch_bounds__(256) void transform_kernel(const float* __restrict__ pos,
                                                        const int* __restrict__ batch,
                                                        const float* __restrict__ pooled,
                                                        float* __restrict__ out) {
    const int n = blockIdx.x * 256 + threadIdx.x;
    const int b = batch[n];
    float T[9];
#pragma unroll
    for (int i = 0; i < 9; ++i) T[i] = pooled[b * 9 + i];
    const float px = pos[n * 3 + 0];
    const float py = pos[n * 3 + 1];
    const float pz = pos[n * 3 + 2];
#pragma unroll
    for (int k = 0; k < 3; ++k)
        out[n * 3 + k] = px * T[0 * 3 + k] + py * T[1 * 3 + k] + pz * T[2 * 3 + k];
}

// ------------------------------- launcher ----------------------------------
extern "C" void kernel_launch(void* const* d_in, const int* in_sizes, int n_in,
                              void* d_out, int out_size, void* d_ws, size_t ws_size,
                              hipStream_t stream) {
    (void)in_sizes; (void)n_in; (void)out_size; (void)ws_size;
    const float* pos   = (const float*)d_in[0];
    const int*   batch = (const int*)d_in[1];
    // d_in[2] = num_graphs (scalar, compile-time constant here)
    const float* W1 = (const float*)d_in[3];
    const float* b1 = (const float*)d_in[4];
    const float* W2 = (const float*)d_in[5];
    const float* b2 = (const float*)d_in[6];
    const float* W3 = (const float*)d_in[7];
    const float* b3 = (const float*)d_in[8];
    float* out = (float*)d_out;

    // Workspace layout (all 256B-aligned, ~29.8 MB total)
    char* ws = (char*)d_ws;
    size_t off = 0;
    int*      nbr    = (int*)(ws + off);      off += (size_t)NPTS * KNN * 4;      // 4.00 MB
    _Float16* w1p    = (_Float16*)(ws + off); off += (size_t)F1 * H1 * 2;         // 0.19 MB
    _Float16* w2p    = (_Float16*)(ws + off); off += (size_t)H1 * H2 * 2;         // 0.50 MB
    _Float16* feat   = (_Float16*)(ws + off); off += (size_t)CHUNK * F1 * 2;      // 1.50 MB
    _Float16* h1buf  = (_Float16*)(ws + off); off += (size_t)CHUNK * H1 * 2;      // 16.0 MB
    _Float16* h2buf  = (_Float16*)(ws + off); off += (size_t)CHUNK * H2 * 2;      // 4.00 MB
    float*    x9     = (float*)(ws + off);    off += (size_t)NPTS * 9 * 4;        // 2.25 MB
    float*    pooled = (float*)(ws + off);    off += (size_t)BGRAPH * 9 * 4;

    // 1) KNN over all points
    knn_kernel<<<NPTS / 256, 256, 0, stream>>>(pos, nbr);

    // 2) One-time f32 -> f16 fragment packing of W1, W2
    pack_b_kernel<<<dim3(H1 / 16, F1 / 32), 32, 0, stream>>>(W1, w1p, F1, H1);
    pack_b_kernel<<<dim3(H2 / 16, H1 / 32), 32, 0, stream>>>(W2, w2p, H1, H2);

    // 3) Chunked MLP pipeline (keeps scratch < 30 MB)
    //    GEMM grids: x = row groups (32 rows), y = col groups of 256 cols
    //    (4 waves/block x 64 cols); exact multiples -> EXEC all-ones for WMMA.
    for (int c = 0; c < NCHUNK; ++c) {
        const int rowStart = c * CHUNK;
        feat_kernel<<<CHUNK / 256, 256, 0, stream>>>(pos, nbr, feat, rowStart);
        wmma_gemm_bias_relu<<<dim3(CHUNK / 32, H1 / 256), 128, 0, stream>>>(
            feat, w1p, b1, h1buf, F1, H1, 1);
        wmma_gemm_bias_relu<<<dim3(CHUNK / 32, H2 / 256), 128, 0, stream>>>(
            h1buf, w2p, b2, h2buf, H1, H2, 1);
        layer3_kernel<<<CHUNK / 256, 256, 0, stream>>>(h2buf, W3, b3, x9, rowStart);
    }

    // 4) Per-graph mean pool + final transform
    pool_kernel<<<BGRAPH, 256, 0, stream>>>(x9, pooled);
    transform_kernel<<<NPTS / 256, 256, 0, stream>>>(pos, batch, pooled, out);
}
